// EuclideanAttention_19207093747780
// MI455X (gfx1250) — matrix operand order repaired and checked by reading, not compiled, not run on hardware
//
#include <hip/hip_runtime.h>
#include <hip/hip_bf16.h>

// EuclideanAttention on gfx1250 (MI455X), all matmuls via v_wmma_f32_16x16x32_bf16.
// Workspace layout (bf16): q[2*16*2048*64], k[same], vT[2*16*64*2048], vals[4096*1024]
// => 4 * 8 MB = 32 MB total.

typedef __attribute__((ext_vector_type(16))) __bf16 v16bf;
typedef __attribute__((ext_vector_type(8)))  __bf16 v8bf;
typedef __attribute__((ext_vector_type(8)))  float  v8f;
typedef __attribute__((ext_vector_type(4)))  float  v4f;

static __device__ inline __bf16 f2bf(float f) { return (__bf16)f; }
static __device__ inline float  bf2f(__bf16 b) { return (float)b; }

// Build a 16-element bf16 fragment from two contiguous 8-element (16B) chunks.
static __device__ inline v16bf load_frag(const __bf16* p0, const __bf16* p1) {
  v8bf a = *(const v8bf*)p0;
  v8bf b = *(const v8bf*)p1;
  v16bf r;
#pragma unroll
  for (int i = 0; i < 8; ++i) { r[i] = a[i]; r[i + 8] = b[i]; }
  return r;
}
static __device__ inline v16bf load_frag16(const __bf16* p) { return load_frag(p, p + 8); }

static __device__ inline v8f wmma_bf16(v16bf a, v16bf b, v8f c) {
  return __builtin_amdgcn_wmma_f32_16x16x32_bf16(false, a, false, b, (short)0, c,
                                                 false, false);
}

// ---------------------------------------------------------------------------
// Kernel 1: qkv = x @ W_qkv + b_qkv, scattered into per-head q / k / vT (bf16).
// Grid: (3072/64, 4096/128); block 256 (8 waves, each a 16x64 output tile).
// ---------------------------------------------------------------------------
__global__ __launch_bounds__(256) void qkv_gemm_kernel(
    const float* __restrict__ x, const float* __restrict__ Wqkv,
    const float* __restrict__ bqkv,
    __bf16* __restrict__ qws, __bf16* __restrict__ kws, __bf16* __restrict__ vtws) {
  __shared__ __align__(16) __bf16 BsT[64][40];  // W tile, transposed [n][k], 80B stride
  const int nbase = blockIdx.x * 64;
  const int mbase = blockIdx.y * 128;
  const int wave = threadIdx.x >> 5;
  const int lane = threadIdx.x & 31;
  const int lidx = lane & 15;
  const int half = lane >> 4;
  const int lo = half * 8;     // unused; A fragment also uses koff
  const int koff = half * 16;  // B-fragment K half offset

  v8f acc[4];
#pragma unroll
  for (int g = 0; g < 4; ++g)
#pragma unroll
    for (int j = 0; j < 8; ++j) acc[g][j] = 0.0f;

  const int myrow = mbase + wave * 16 + lidx;
  const float* arow = x + (size_t)myrow * 1024;

  for (int kb = 0; kb < 1024; kb += 32) {
    __syncthreads();
    {  // cooperatively stage W tile (32 x 64 f32) into LDS as bf16, transposed
      const int t = threadIdx.x;
      const int krow = t >> 3;
      const int nc0 = (t & 7) * 8;
      const float* src = Wqkv + (size_t)(kb + krow) * 3072 + nbase + nc0;
      v4f f0 = *(const v4f*)src;
      v4f f1 = *(const v4f*)(src + 4);
#pragma unroll
      for (int j = 0; j < 4; ++j) BsT[nc0 + j][krow] = f2bf(f0[j]);
#pragma unroll
      for (int j = 0; j < 4; ++j) BsT[nc0 + 4 + j][krow] = f2bf(f1[j]);
    }
    __syncthreads();

    v16bf a;  // A fragment: row = lidx, K = {koff..koff+15}
#pragma unroll
    for (int i = 0; i < 8; ++i) a[i] = f2bf(arow[kb + koff + i]);
#pragma unroll
    for (int i = 0; i < 8; ++i) a[8 + i] = f2bf(arow[kb + koff + 8 + i]);

    // Hoist all 4 B-fragment LDS loads ahead of the 4 WMMAs.
    v16bf bfrag[4];
#pragma unroll
    for (int g = 0; g < 4; ++g) bfrag[g] = load_frag16(&BsT[g * 16 + lidx][koff]);
#pragma unroll
    for (int g = 0; g < 4; ++g) acc[g] = wmma_bf16(a, bfrag[g], acc[g]);
  }

  // Epilogue: bias + scatter into q / k / vT. A 16-col group never straddles a
  // q/k/v or head boundary (192 = 12*16), so h / region / d-subgroup are
  // uniform per group -> scalar branches, no lane divergence.
#pragma unroll
  for (int g = 0; g < 4; ++g) {
    const int gi = (nbase >> 4) + g;   // global 16-column group index
    const int h  = gi / 12;            // head (uniform)
    const int gr = gi % 12;            // group within head
    const int r  = gr >> 2;            // 0=q 1=k 2=v (uniform)
    const int d  = (gr & 3) * 16 + lidx;
    const float bias = bqkv[nbase + g * 16 + lidx];
#pragma unroll
    for (int j = 0; j < 8; ++j) {
      const int m = mbase + wave * 16 + j + half * 8;  // C layout: row = j + 8*half
      const int b_ = m >> 11;
      const int s = m & 2047;
      const int bh = b_ * 16 + h;
      const float v = acc[g][j] + bias;
      if (r == 0)
        qws[((size_t)(bh * 2048 + s)) * 64 + d] = f2bf(v);
      else if (r == 1)
        kws[((size_t)(bh * 2048 + s)) * 64 + d] = f2bf(v);
      else
        vtws[((size_t)(bh * 64 + d)) * 2048 + s] = f2bf(v);
    }
  }
}

// ---------------------------------------------------------------------------
// Kernel 2: flash attention with logits = 2*qk - |k|^2 (q_sq drops out of
// softmax). One wave = 16 query rows of one (b,h); loop over 64 tiles of 32 keys.
// Grid: 512 blocks x 8 waves = 32 heads * 128 row-tiles.
// ---------------------------------------------------------------------------
__global__ __launch_bounds__(256) void attn_kernel(
    const __bf16* __restrict__ qws, const __bf16* __restrict__ kws,
    const __bf16* __restrict__ vtws, __bf16* __restrict__ vals) {
  __shared__ __align__(16) __bf16 plds[8][16][32];  // wave-private P staging
  const int wave = threadIdx.x >> 5;
  const int lane = threadIdx.x & 31;
  const int lidx = lane & 15;
  const int half = lane >> 4;
  const int lo = half * 8;
  const int koff = half * 16;
  const int gw = blockIdx.x * 8 + wave;
  const int bh = gw >> 7;
  const int rt = gw & 127;

  const __bf16* qh = qws + (size_t)bh * 2048 * 64;
  const __bf16* kh = kws + (size_t)bh * 2048 * 64;
  const __bf16* vh = vtws + (size_t)bh * 64 * 2048;

  const __bf16* qrow = qh + (size_t)(rt * 16 + lidx) * 64;
  const v16bf qa0 = load_frag16(qrow + koff);        // dims 0..31
  const v16bf qa1 = load_frag16(qrow + 32 + koff);   // dims 32..63

  float Mr[8], Lr[8];
  v8f acc[4];
#pragma unroll
  for (int j = 0; j < 8; ++j) { Mr[j] = -3.0e38f; Lr[j] = 0.0f; }
#pragma unroll
  for (int g = 0; g < 4; ++g)
#pragma unroll
    for (int j = 0; j < 8; ++j) acc[g][j] = 0.0f;

  for (int kt = 0; kt < 64; ++kt) {
    const int kbase = kt * 32;
    // K^T B-fragments: lane = key column, contiguous 16 dims per lane half
    const __bf16* kr0 = kh + (size_t)(kbase + lidx) * 64;
    const __bf16* kr1 = kh + (size_t)(kbase + 16 + lidx) * 64;
    v16bf kb00 = load_frag16(kr0 + koff);
    v16bf kb01 = load_frag16(kr0 + 32 + koff);
    v16bf kb10 = load_frag16(kr1 + koff);
    v16bf kb11 = load_frag16(kr1 + 32 + koff);

    // |k|^2 per key column: lane-local partial over 32 dims + cross-half add
    float ks0 = 0.0f, ks1 = 0.0f;
#pragma unroll
    for (int i = 0; i < 16; ++i) {
      float f;
      f = bf2f(kb00[i]); ks0 += f * f;
      f = bf2f(kb01[i]); ks0 += f * f;
      f = bf2f(kb10[i]); ks1 += f * f;
      f = bf2f(kb11[i]); ks1 += f * f;
    }
    ks0 += __shfl_xor(ks0, 16, 32);
    ks1 += __shfl_xor(ks1, 16, 32);

    v8f qk0, qk1;
#pragma unroll
    for (int j = 0; j < 8; ++j) { qk0[j] = 0.0f; qk1[j] = 0.0f; }
    qk0 = wmma_bf16(qa0, kb00, qk0);
    qk0 = wmma_bf16(qa1, kb01, qk0);
    qk1 = wmma_bf16(qa0, kb10, qk1);
    qk1 = wmma_bf16(qa1, kb11, qk1);

    // Online softmax; row j lives across the 16 lanes of this half.
    float p0[8], p1[8], alpha[8];
#pragma unroll
    for (int j = 0; j < 8; ++j) {
      float l0 = 2.0f * qk0[j] - ks0;
      float l1 = 2.0f * qk1[j] - ks1;
      float mx = fmaxf(l0, l1);
#pragma unroll
      for (int m = 1; m < 16; m <<= 1) mx = fmaxf(mx, __shfl_xor(mx, m, 32));
      const float mnew = fmaxf(Mr[j], mx);
      alpha[j] = __expf(Mr[j] - mnew);
      Mr[j] = mnew;
      p0[j] = __expf(l0 - mnew);
      p1[j] = __expf(l1 - mnew);
      float rs = p0[j] + p1[j];
#pragma unroll
      for (int m = 1; m < 16; m <<= 1) rs += __shfl_xor(rs, m, 32);
      Lr[j] = Lr[j] * alpha[j] + rs;
    }
#pragma unroll
    for (int g = 0; g < 4; ++g)
#pragma unroll
      for (int j = 0; j < 8; ++j) acc[g][j] *= alpha[j];

    // Re-shape P from C/D layout to A layout through wave-private LDS.
#pragma unroll
    for (int j = 0; j < 8; ++j) {
      plds[wave][j + half * 8][lidx] = f2bf(p0[j]);
      plds[wave][j + half * 8][lidx + 16] = f2bf(p1[j]);
    }
    asm volatile("s_wait_dscnt 0" ::: "memory");
    const __bf16* pr = &plds[wave][lidx][0];
    v16bf pa = load_frag16(pr + koff);

    // O += P x V : hoist all 4 V B-fragment loads, then 4 back-to-back WMMAs.
    v16bf vb[4];
#pragma unroll
    for (int g = 0; g < 4; ++g)
      vb[g] = load_frag16(vh + (size_t)(g * 16 + lidx) * 2048 + kbase + koff);
#pragma unroll
    for (int g = 0; g < 4; ++g) acc[g] = wmma_bf16(pa, vb[g], acc[g]);
  }

  // Epilogue: divide by row sums, store vals[b, s, h*64 + d] (bf16)
  const int b_ = bh >> 4;
  const int h = bh & 15;
#pragma unroll
  for (int j = 0; j < 8; ++j) {
    const float inv = 1.0f / Lr[j];
    const int s = rt * 16 + j + half * 8;
#pragma unroll
    for (int g = 0; g < 4; ++g) {
      vals[((size_t)(b_ * 2048 + s)) * 1024 + h * 64 + g * 16 + lidx] =
          f2bf(acc[g][j] * inv);
    }
  }
}

// ---------------------------------------------------------------------------
// Kernel 3: out = vals @ W_o + b_o  (M=4096, N=1024, K=1024), f32 out.
// ---------------------------------------------------------------------------
__global__ __launch_bounds__(256) void oproj_kernel(
    const __bf16* __restrict__ vals, const float* __restrict__ Wo,
    const float* __restrict__ bo, float* __restrict__ out) {
  __shared__ __align__(16) __bf16 BsT[64][40];
  const int nbase = blockIdx.x * 64;
  const int mbase = blockIdx.y * 128;
  const int wave = threadIdx.x >> 5;
  const int lane = threadIdx.x & 31;
  const int lidx = lane & 15;
  const int half = lane >> 4;
  const int lo = half * 8;
  const int koff = half * 16;

  v8f acc[4];
#pragma unroll
  for (int g = 0; g < 4; ++g)
#pragma unroll
    for (int j = 0; j < 8; ++j) acc[g][j] = 0.0f;

  const int myrow = mbase + wave * 16 + lidx;
  const __bf16* arow = vals + (size_t)myrow * 1024;

  for (int kb = 0; kb < 1024; kb += 32) {
    __syncthreads();
    {
      const int t = threadIdx.x;
      const int krow = t >> 3;
      const int nc0 = (t & 7) * 8;
      const float* src = Wo + (size_t)(kb + krow) * 1024 + nbase + nc0;
      v4f f0 = *(const v4f*)src;
      v4f f1 = *(const v4f*)(src + 4);
#pragma unroll
      for (int j = 0; j < 4; ++j) BsT[nc0 + j][krow] = f2bf(f0[j]);
#pragma unroll
      for (int j = 0; j < 4; ++j) BsT[nc0 + 4 + j][krow] = f2bf(f1[j]);
    }
    __syncthreads();

    v16bf a = load_frag16(arow + kb + koff);
    v16bf bfrag[4];
#pragma unroll
    for (int g = 0; g < 4; ++g) bfrag[g] = load_frag16(&BsT[g * 16 + lidx][koff]);
#pragma unroll
    for (int g = 0; g < 4; ++g) acc[g] = wmma_bf16(a, bfrag[g], acc[g]);
  }

#pragma unroll
  for (int g = 0; g < 4; ++g) {
    const int n = nbase + g * 16 + lidx;
    const float bias = bo[n];
#pragma unroll
    for (int j = 0; j < 8; ++j) {
      const int m = mbase + wave * 16 + j + half * 8;
      out[(size_t)m * 1024 + n] = acc[g][j] + bias;
    }
  }
}

// ---------------------------------------------------------------------------
extern "C" void kernel_launch(void* const* d_in, const int* in_sizes, int n_in,
                              void* d_out, int out_size, void* d_ws, size_t ws_size,
                              hipStream_t stream) {
  const float* x    = (const float*)d_in[0];   // [2, 2048, 1024]
  const float* Wqkv = (const float*)d_in[1];   // [1024, 3072]
  const float* bqkv = (const float*)d_in[2];   // [3072]
  const float* Wo   = (const float*)d_in[3];   // [1024, 1024]
  const float* bo   = (const float*)d_in[4];   // [1024]
  float* out = (float*)d_out;                  // [2, 2048, 1024] f32

  const size_t HEAD_ELEMS = (size_t)2 * 16 * 2048 * 64;  // 4 Mi elems per tensor
  __bf16* qws  = (__bf16*)d_ws;
  __bf16* kws  = qws + HEAD_ELEMS;
  __bf16* vtws = kws + HEAD_ELEMS;
  __bf16* vals = vtws + HEAD_ELEMS;            // [4096, 1024] bf16

  qkv_gemm_kernel<<<dim3(48, 32), 256, 0, stream>>>(x, Wqkv, bqkv, qws, kws, vtws);
  attn_kernel<<<dim3(512), 256, 0, stream>>>(qws, kws, vtws, vals);
  oproj_kernel<<<dim3(16, 32), 256, 0, stream>>>(vals, Wo, bo, out);
}
